// EncoderDecoderCNNLSTM_41145786696008
// MI455X (gfx1250) — matrix operand order repaired and checked by reading, not compiled, not run on hardware
//
#include <hip/hip_runtime.h>
#include <hip/hip_bf16.h>

// ---------------------------------------------------------------------------
// Types for CDNA5 WMMA (wave32): v_wmma_f32_16x16x32_bf16
// ---------------------------------------------------------------------------
typedef __attribute__((ext_vector_type(16))) __bf16 v16bf;
typedef __attribute__((ext_vector_type(8)))  float  v8f;

union Frag {
    v16bf v;
    uint4 u[2];   // 32 bytes = 16 bf16
};

__device__ __forceinline__ unsigned short f2bf(float f) {
    unsigned int u = __float_as_uint(f);
    unsigned int r = (u + 0x7FFFu + ((u >> 16) & 1u)) >> 16;
    return (unsigned short)r;
}
__device__ __forceinline__ float bf2f(unsigned short h) {
    return __uint_as_float(((unsigned int)h) << 16);
}
__device__ __forceinline__ float sigm(float v) {
    return 1.0f / (1.0f + __expf(-v));
}

// ---------------------------------------------------------------------------
// Fused LSTM step: gates = bias + X·Wihᵀ + Hprev·Whhᵀ  (bf16 WMMA, f32 acc)
// followed by in-register LSTM pointwise. Each wave owns a 16x16 (batch x j)
// tile and 4 gate accumulators at columns {j, j+512, j+1024, j+1536}.
// Grid: (B/128, 512/16) = (4, 32), block = 256 threads (8 waves along M).
// ---------------------------------------------------------------------------
__global__ __launch_bounds__(256) void lstm_step_kernel(
    const unsigned short* __restrict__ X,      // B rows, ldx elems, K1 valid
    long ldx, int K1,
    const unsigned short* __restrict__ Hprev,  // B x 512 bf16
    const unsigned short* __restrict__ Wih,    // 2048 x K1 row-major bf16
    const unsigned short* __restrict__ Whh,    // 2048 x 512 row-major bf16
    const float* __restrict__ bias,            // 2048  (bih + bhh)
    float* __restrict__ Cst,                   // B x 512 f32, in/out (in place)
    unsigned short* __restrict__ Hout)         // B x 512 bf16 (ping-pong)
{
    const int H = 512;
    const int lane    = threadIdx.x & 31;
    const int wave    = threadIdx.x >> 5;
    const int mBase   = blockIdx.x * 128 + wave * 16;   // batch-row tile
    const int jb      = blockIdx.y * 16;                // hidden-unit tile
    const int halfSel = lane >> 4;                      // 0 | 1
    const int rowSel  = lane & 15;                      // M (for A) / N (for B/C)

    // Accumulators: i, f, g, o. Bias depends only on the column.
    v8f acc[4];
#pragma unroll
    for (int g = 0; g < 4; ++g) {
        float bs = bias[g * H + jb + rowSel];
#pragma unroll
        for (int r = 0; r < 8; ++r) acc[g][r] = bs;
    }

    const int kOffA = halfSel * 8;    // A frag: K runs {off..off+7, off+16..off+23}
    const int kOffB = halfSel * 8;    // B frag: K runs {off..off+7, off+16..off+23}

    // ---- Phase 1: X · Wihᵀ -------------------------------------------------
    {
        const unsigned short* aRow = X + (long)(mBase + rowSel) * ldx;
        for (int kb = 0; kb < K1; kb += 32) {
            Frag a;
            a.u[0] = *(const uint4*)(aRow + kb + kOffA);
            a.u[1] = *(const uint4*)(aRow + kb + kOffA + 16);
            __builtin_prefetch(aRow + kb + 32, 0, 1);
#pragma unroll
            for (int g = 0; g < 4; ++g) {
                const unsigned short* bRow = Wih + (long)(g * H + jb + rowSel) * K1;
                Frag b;
                b.u[0] = *(const uint4*)(bRow + kb + kOffB);
                b.u[1] = *(const uint4*)(bRow + kb + kOffB + 16);
                acc[g] = __builtin_amdgcn_wmma_f32_16x16x32_bf16(
                    false, a.v, false, b.v, (short)0, acc[g], false, false);
            }
        }
    }
    // ---- Phase 2: Hprev · Whhᵀ --------------------------------------------
    {
        const unsigned short* aRow = Hprev + (long)(mBase + rowSel) * H;
        for (int kb = 0; kb < H; kb += 32) {
            Frag a;
            a.u[0] = *(const uint4*)(aRow + kb + kOffA);
            a.u[1] = *(const uint4*)(aRow + kb + kOffA + 16);
            __builtin_prefetch(aRow + kb + 32, 0, 1);
#pragma unroll
            for (int g = 0; g < 4; ++g) {
                const unsigned short* bRow = Whh + (long)(g * H + jb + rowSel) * H;
                Frag b;
                b.u[0] = *(const uint4*)(bRow + kb + kOffB);
                b.u[1] = *(const uint4*)(bRow + kb + kOffB + 16);
                acc[g] = __builtin_amdgcn_wmma_f32_16x16x32_bf16(
                    false, a.v, false, b.v, (short)0, acc[g], false, false);
            }
        }
    }

    // ---- LSTM pointwise, in-register. C/D layout: vgpr r -> M = r + 8*half,
    // lane -> N = lane & 15. Each (b,j) owned by exactly one lane => c in place.
    const int j = jb + rowSel;
#pragma unroll
    for (int r = 0; r < 8; ++r) {
        int  b   = mBase + r + (halfSel << 3);
        long idx = (long)b * H + j;
        float iv = sigm(acc[0][r]);
        float fv = sigm(acc[1][r]);
        float gv = tanhf(acc[2][r]);
        float ov = sigm(acc[3][r]);
        float cn = fv * Cst[idx] + iv * gv;
        Cst[idx]  = cn;
        Hout[idx] = f2bf(ov * tanhf(cn));
    }
}

// ---------------------------------------------------------------------------
// Conv1d (B,336,16) -> (B,336,128), K=5, pad=2, writes bf16 for WMMA consumption
// ---------------------------------------------------------------------------
__global__ void conv1d_kernel(const float* __restrict__ x,      // (512,384,16)
                              const float* __restrict__ w,      // (128,16,5)
                              const float* __restrict__ bc,     // (128)
                              unsigned short* __restrict__ out) // (512,336,128)
{
    long idx = (long)blockIdx.x * blockDim.x + threadIdx.x;
    if (idx >= (long)512 * 336 * 128) return;
    int co = (int)(idx & 127);
    int t  = (int)((idx >> 7) % 336);
    int b  = (int)(idx / (336 * 128));
    float s = bc[co];
    const float* wr = w + co * 80;
#pragma unroll
    for (int k = 0; k < 5; ++k) {
        int tt = t + k - 2;
        if (tt < 0 || tt >= 336) continue;
        const float* xr = x + ((long)b * 384 + tt) * 16;
#pragma unroll
        for (int ci = 0; ci < 16; ++ci) s += xr[ci] * wr[ci * 5 + k];
    }
    out[idx] = f2bf(s);
}

// ---------------------------------------------------------------------------
// Prep / small kernels
// ---------------------------------------------------------------------------
__global__ void f32_to_bf16_kernel(const float* __restrict__ in,
                                   unsigned short* __restrict__ out, long n) {
    long i = (long)blockIdx.x * blockDim.x + threadIdx.x;
    if (i < n) out[i] = f2bf(in[i]);
}
__global__ void pad_wih_dec0_kernel(const float* __restrict__ in,     // 2048x16
                                    unsigned short* __restrict__ out) // 2048x32
{
    int i = blockIdx.x * blockDim.x + threadIdx.x;
    if (i >= 2048 * 32) return;
    int j = i & 31, r = i >> 5;
    out[i] = (j < 16) ? f2bf(in[r * 16 + j]) : (unsigned short)0;
}
__global__ void bias_sum_kernel(const float* a, const float* b, float* o, int n) {
    int i = blockIdx.x * blockDim.x + threadIdx.x;
    if (i < n) o[i] = a[i] + b[i];
}
__global__ void fill_zero_u16(unsigned short* p, long n) {
    long i = (long)blockIdx.x * blockDim.x + threadIdx.x;
    if (i < n) p[i] = 0;
}
__global__ void fill_zero_f32(float* p, long n) {
    long i = (long)blockIdx.x * blockDim.x + threadIdx.x;
    if (i < n) p[i] = 0.0f;
}

// Decoder input build: din[b, 0..15] from x[:, 335+t, :] with feature 0
// replaced by previous (dropout-scaled) output when t>0; pad 16..31 with 0.
__global__ void dec_input_kernel(const float* __restrict__ x,
                                 const float* __restrict__ out_prev,
                                 unsigned short* __restrict__ din, int t)
{
    int i = blockIdx.x * blockDim.x + threadIdx.x;
    if (i >= 512 * 32) return;
    int j = i & 31, b = i >> 5;
    float v = 0.0f;
    if (j < 16) {
        if (j == 0 && t > 0) v = out_prev[b];
        else v = x[((long)b * 384 + (335 + t)) * 16 + j];
    }
    din[i] = f2bf(v);
}

// FC head + deterministic pseudo-dropout. One wave per batch row.
__global__ __launch_bounds__(256) void fc_out_kernel(
    const unsigned short* __restrict__ h1, const float* __restrict__ fcw,
    const float* __restrict__ fcb, float* __restrict__ out,
    float* __restrict__ out_prev, int t)
{
    int lane = threadIdx.x & 31;
    int b    = blockIdx.x * 8 + (threadIdx.x >> 5);
    const unsigned short* hr = h1 + (long)b * 512;
    float s = 0.0f;
    for (int k = lane; k < 512; k += 32) s += bf2f(hr[k]) * fcw[k];
#pragma unroll
    for (int off = 16; off > 0; off >>= 1) s += __shfl_xor(s, off, 32);
    if (lane == 0) {
        float o = s + fcb[0];
        unsigned h = (unsigned)(t * 0x9E3779B9u) ^ (unsigned)(b * 0x85EBCA6Bu);
        h ^= h >> 16; h *= 0x7FEB352Du; h ^= h >> 15; h *= 0x846CA68Bu; h ^= h >> 16;
        float val = ((h % 100u) >= 5u) ? o * (1.0f / 0.95f) : 0.0f;
        out[(long)b * 48 + t] = val;
        out_prev[b] = val;
    }
}

// ---------------------------------------------------------------------------
// Host orchestration
// ---------------------------------------------------------------------------
extern "C" void kernel_launch(void* const* d_in, const int* in_sizes, int n_in,
                              void* d_out, int out_size, void* d_ws, size_t ws_size,
                              hipStream_t stream) {
    (void)in_sizes; (void)n_in; (void)out_size; (void)ws_size;
    const float* x       = (const float*)d_in[0];
    const float* conv_w  = (const float*)d_in[1];
    const float* conv_b  = (const float*)d_in[2];
    const float* e_wih0  = (const float*)d_in[3];
    const float* e_whh0  = (const float*)d_in[4];
    const float* e_bih0  = (const float*)d_in[5];
    const float* e_bhh0  = (const float*)d_in[6];
    const float* e_wih1  = (const float*)d_in[7];
    const float* e_whh1  = (const float*)d_in[8];
    const float* e_bih1  = (const float*)d_in[9];
    const float* e_bhh1  = (const float*)d_in[10];
    const float* d_wih0  = (const float*)d_in[11];
    const float* d_whh0  = (const float*)d_in[12];
    const float* d_bih0  = (const float*)d_in[13];
    const float* d_bhh0  = (const float*)d_in[14];
    const float* d_wih1  = (const float*)d_in[15];
    const float* d_whh1  = (const float*)d_in[16];
    const float* d_bih1  = (const float*)d_in[17];
    const float* d_bhh1  = (const float*)d_in[18];
    const float* fc_w    = (const float*)d_in[19];
    const float* fc_b    = (const float*)d_in[20];
    float* out = (float*)d_out;

    const long B = 512, H = 512, G = 2048, T = 336;

    char* ws = (char*)d_ws;
    size_t off = 0;
    auto carve = [&](size_t bytes) -> char* {
        char* p = ws + off;
        off = (off + bytes + 255) & ~(size_t)255;
        return p;
    };
    unsigned short* wih0e = (unsigned short*)carve(G * 128 * 2);
    unsigned short* whh0e = (unsigned short*)carve(G * H * 2);
    unsigned short* wih1e = (unsigned short*)carve(G * H * 2);
    unsigned short* whh1e = (unsigned short*)carve(G * H * 2);
    unsigned short* wih0d = (unsigned short*)carve(G * 32 * 2);
    unsigned short* whh0d = (unsigned short*)carve(G * H * 2);
    unsigned short* wih1d = (unsigned short*)carve(G * H * 2);
    unsigned short* whh1d = (unsigned short*)carve(G * H * 2);
    float* be0 = (float*)carve(G * 4);
    float* be1 = (float*)carve(G * 4);
    float* bd0 = (float*)carve(G * 4);
    float* bd1 = (float*)carve(G * 4);
    unsigned short* convc = (unsigned short*)carve(B * T * 128 * 2);
    unsigned short* h0p[2], * h1p[2];
    h0p[0] = (unsigned short*)carve(B * H * 2);
    h0p[1] = (unsigned short*)carve(B * H * 2);
    h1p[0] = (unsigned short*)carve(B * H * 2);
    h1p[1] = (unsigned short*)carve(B * H * 2);
    float* c0 = (float*)carve(B * H * 4);
    float* c1 = (float*)carve(B * H * 4);
    unsigned short* din = (unsigned short*)carve(B * 32 * 2);
    float* oprev = (float*)carve(B * 4);

    const int TPB = 256;
    auto blk = [](long n) { return (unsigned)((n + 255) / 256); };

    // Weight/bias prep (bf16 conversion; decoder L0 Wih zero-padded K 16->32)
    f32_to_bf16_kernel<<<blk(G * 128), TPB, 0, stream>>>(e_wih0, wih0e, G * 128);
    f32_to_bf16_kernel<<<blk(G * H), TPB, 0, stream>>>(e_whh0, whh0e, G * H);
    f32_to_bf16_kernel<<<blk(G * H), TPB, 0, stream>>>(e_wih1, wih1e, G * H);
    f32_to_bf16_kernel<<<blk(G * H), TPB, 0, stream>>>(e_whh1, whh1e, G * H);
    pad_wih_dec0_kernel<<<blk(G * 32), TPB, 0, stream>>>(d_wih0, wih0d);
    f32_to_bf16_kernel<<<blk(G * H), TPB, 0, stream>>>(d_whh0, whh0d, G * H);
    f32_to_bf16_kernel<<<blk(G * H), TPB, 0, stream>>>(d_wih1, wih1d, G * H);
    f32_to_bf16_kernel<<<blk(G * H), TPB, 0, stream>>>(d_whh1, whh1d, G * H);
    bias_sum_kernel<<<blk(G), TPB, 0, stream>>>(e_bih0, e_bhh0, be0, (int)G);
    bias_sum_kernel<<<blk(G), TPB, 0, stream>>>(e_bih1, e_bhh1, be1, (int)G);
    bias_sum_kernel<<<blk(G), TPB, 0, stream>>>(d_bih0, d_bhh0, bd0, (int)G);
    bias_sum_kernel<<<blk(G), TPB, 0, stream>>>(d_bih1, d_bhh1, bd1, (int)G);

    // Zero initial states
    fill_zero_u16<<<blk(B * H), TPB, 0, stream>>>(h0p[0], B * H);
    fill_zero_u16<<<blk(B * H), TPB, 0, stream>>>(h1p[0], B * H);
    fill_zero_f32<<<blk(B * H), TPB, 0, stream>>>(c0, B * H);
    fill_zero_f32<<<blk(B * H), TPB, 0, stream>>>(c1, B * H);

    // Conv front-end
    conv1d_kernel<<<blk(B * T * 128), TPB, 0, stream>>>(x, conv_w, conv_b, convc);

    dim3 sgrid(4, 32);  // (B/128, H/16)

    // ---- Encoder: 336 steps, two stacked layers, h ping-pong ---------------
    for (int t = 0; t < 336; ++t) {
        int p = t & 1;
        lstm_step_kernel<<<sgrid, TPB, 0, stream>>>(
            convc + (long)t * 128, (long)T * 128, 128,
            h0p[p], wih0e, whh0e, be0, c0, h0p[p ^ 1]);
        lstm_step_kernel<<<sgrid, TPB, 0, stream>>>(
            h0p[p ^ 1], H, (int)H,
            h1p[p], wih1e, whh1e, be1, c1, h1p[p ^ 1]);
    }

    // ---- Decoder: 48 autoregressive steps (states carry over, 336 even => p=0)
    int p = 0;
    for (int t = 0; t < 48; ++t) {
        dec_input_kernel<<<blk(B * 32), TPB, 0, stream>>>(x, oprev, din, t);
        lstm_step_kernel<<<sgrid, TPB, 0, stream>>>(
            din, 32, 32, h0p[p], wih0d, whh0d, bd0, c0, h0p[p ^ 1]);
        lstm_step_kernel<<<sgrid, TPB, 0, stream>>>(
            h0p[p ^ 1], H, (int)H, h1p[p], wih1d, whh1d, bd1, c1, h1p[p ^ 1]);
        fc_out_kernel<<<64, TPB, 0, stream>>>(h1p[p ^ 1], fc_w, fc_b, out, oprev, t);
        p ^= 1;
    }
}
